// SelfAttention_31104153157916
// MI455X (gfx1250) — matrix operand order repaired and checked by reading, not compiled, not run on hardware
//
#include <hip/hip_runtime.h>
#include <hip/hip_bf16.h>

// ---------------- problem constants (fixed by the reference) ----------------
constexpr int BZ = 8;      // batch
constexpr int SQ = 2048;   // sequence length
constexpr int ED = 1024;   // embedding dim
constexpr int DD = 1024;   // projection dim
#define NEGV (-1e20f)
// scale = sqrt(1024) + float32 eps
#define INV_SCALE (1.0f / (32.0f + 1.1920928955078125e-07f))

typedef __bf16 bf16_t;
typedef __attribute__((ext_vector_type(16))) __bf16 v16bf;
typedef __attribute__((ext_vector_type(8)))  __bf16 v8bf;
typedef __attribute__((ext_vector_type(4)))  __bf16 v4bf;
typedef __attribute__((ext_vector_type(8)))  float  v8f;
typedef __attribute__((ext_vector_type(4)))  float  v4f;

// ---------------- WMMA helpers (CDNA5 16x16x32 bf16, f32 accum) -------------
__device__ __forceinline__ v8f wmma_bf16(v16bf a, v16bf b, v8f c) {
  // (neg_a, A, neg_b, B, c_mod, C, reuse_a, reuse_b)
  return __builtin_amdgcn_wmma_f32_16x16x32_bf16(false, a, false, b,
                                                 (short)0, c, false, false);
}

// A fragment (16x32, row = lane&15): per lane K = off+{0..7} and off+{16..23},
// off = (lane&16)?8:0.  base is wave-uniform -> saddr-form loads.
__device__ __forceinline__ v16bf load_a_frag(const bf16_t* __restrict__ base,
                                             int off) {
  v8bf lo = *(const v8bf*)(base + off);
  v8bf hi = *(const v8bf*)(base + off + 16);
  v16bf r;
#pragma unroll
  for (int i = 0; i < 8; ++i) { r[i] = lo[i]; r[i + 8] = hi[i]; }
  return r;
}
// B fragment (32x16, col = lane&15): per lane K = off+{0..7} and off+{16..23},
// off = (lane&16)?8:0 (same K order as the A fragment).
__device__ __forceinline__ v16bf load_b_frag(const bf16_t* __restrict__ base,
                                             int off) {
  v8bf lo = *(const v8bf*)(base + off);
  v8bf hi = *(const v8bf*)(base + off + 16);
  v16bf r;
#pragma unroll
  for (int i = 0; i < 8; ++i) { r[i] = lo[i]; r[i + 8] = hi[i]; }
  return r;
}

// ---------------- precision conversion kernels ------------------------------
__global__ void __launch_bounds__(256)
cvt_bf16_kernel(const float* __restrict__ in, bf16_t* __restrict__ out, int n) {
  int i = (blockIdx.x * blockDim.x + threadIdx.x) * 4;
  if (i >= n) return;
  v4f x = *(const v4f*)(in + i);
  v4bf y;
#pragma unroll
  for (int j = 0; j < 4; ++j) y[j] = (bf16_t)x[j];
  *(v4bf*)(out + i) = y;
}

// WT[n][k] = W[k][n], fp32 -> bf16  (tiny: 1M elements per weight)
__global__ void __launch_bounds__(256)
transpose_w_kernel(const float* __restrict__ W, bf16_t* __restrict__ WT) {
  int i = blockIdx.x * blockDim.x + threadIdx.x;
  if (i >= ED * DD) return;
  int k = i / DD, n = i - k * DD;
  WT[n * ED + k] = (bf16_t)W[i];
}

// ---------------- QKV projection GEMM ---------------------------------------
// C[m,n] = sum_k A[m,k] * WT[n,k];  64x64 tile per wave (4x4 WMMA tiles).
// TRANS_OUT=false: C row-major [BZ*SQ, DD]  (Q, K)
// TRANS_OUT=true : C = V^T  [BZ][DD][SQ]    (V)
template <bool TRANS_OUT>
__global__ void __launch_bounds__(256)
proj_gemm_kernel(const bf16_t* __restrict__ A,
                 const bf16_t* __restrict__ WT,
                 bf16_t* __restrict__ C) {
  const int lane = threadIdx.x;
  const int m0 = (blockIdx.y * 8 + threadIdx.y) * 64;
  const int n0 = blockIdx.x * 64;
  const int l15 = lane & 15;
  const int hi16 = (lane >> 4) & 1;

  int aoffs[4], boffs[4];                 // 32-bit per-lane element offsets
#pragma unroll
  for (int t = 0; t < 4; ++t) {
    aoffs[t] = (m0 + t * 16 + l15) * ED + hi16 * 8;
    boffs[t] = (n0 + t * 16 + l15) * ED + hi16 * 8;
  }

  v8f acc[4][4] = {};
#pragma unroll 1
  for (int k0 = 0; k0 < ED; k0 += 32) {
    if (k0 + 32 < ED) {  // prefetch next K-chunk (global_prefetch_b8)
      __builtin_prefetch(A + aoffs[0] + k0 + 32, 0, 1);
      __builtin_prefetch(WT + boffs[0] + k0 + 32, 0, 1);
    }
    v16bf bfr[4];
#pragma unroll
    for (int t = 0; t < 4; ++t) bfr[t] = load_b_frag(WT, boffs[t] + k0);
#pragma unroll
    for (int ti = 0; ti < 4; ++ti) {      // one A fragment live at a time
      v16bf a = load_a_frag(A, aoffs[ti] + k0);
#pragma unroll
      for (int tj = 0; tj < 4; ++tj)
        acc[ti][tj] = wmma_bf16(a, bfr[tj], acc[ti][tj]);
    }
  }

  const int rsel = hi16 * 8;              // C layout: lanes 16..31 hold M+8
  // batch index is tile-uniform (tiles never straddle a batch: 2048 % 64 == 0)
  bf16_t* __restrict__ Cb =
      TRANS_OUT ? (C + (size_t)(m0 >> 11) * DD * SQ) : C;
#pragma unroll
  for (int ti = 0; ti < 4; ++ti)
#pragma unroll
    for (int tj = 0; tj < 4; ++tj)
#pragma unroll
      for (int j = 0; j < 8; ++j) {
        int m = m0 + ti * 16 + j + rsel;
        int n = n0 + tj * 16 + l15;
        bf16_t v = (bf16_t)acc[ti][tj][j];
        if (TRANS_OUT) {
          Cb[n * SQ + (m & (SQ - 1))] = v;
        } else {
          Cb[m * DD + n] = v;
        }
      }
}

// ---------------- energy = (Q K^T) with causal+pad mask, scaled -------------
// B operand columns of energy are rows of K (contiguous in D) -> direct loads.
__global__ void __launch_bounds__(256)
energy_kernel(const bf16_t* __restrict__ Q,
              const bf16_t* __restrict__ Kmat,
              const int* __restrict__ mask,
              float* __restrict__ Eo) {
  const int lane = threadIdx.x;
  const int b = blockIdx.z;
  const int q0 = (blockIdx.y * 8 + threadIdx.y) * 64;
  const int k0c = blockIdx.x * 64;
  const int l15 = lane & 15;
  const int hi16 = (lane >> 4) & 1;

  const bf16_t* __restrict__ Qb = Q    + (size_t)b * SQ * DD;  // uniform bases
  const bf16_t* __restrict__ Kb = Kmat + (size_t)b * SQ * DD;

  v8f acc[4][4] = {};
  if (k0c <= q0 + 63) {  // tiles fully above the diagonal skip all math
    int aoffs[4], boffs[4];
#pragma unroll
    for (int t = 0; t < 4; ++t) {
      aoffs[t] = (q0  + t * 16 + l15) * DD + hi16 * 8;
      boffs[t] = (k0c + t * 16 + l15) * DD + hi16 * 8;
    }
#pragma unroll 1
    for (int k0 = 0; k0 < DD; k0 += 32) {
      if (k0 + 32 < DD) {
        __builtin_prefetch(Qb + aoffs[0] + k0 + 32, 0, 1);
        __builtin_prefetch(Kb + boffs[0] + k0 + 32, 0, 1);
      }
      v16bf bfr[4];
#pragma unroll
      for (int t = 0; t < 4; ++t) bfr[t] = load_b_frag(Kb, boffs[t] + k0);
#pragma unroll
      for (int ti = 0; ti < 4; ++ti) {
        v16bf a = load_a_frag(Qb, aoffs[ti] + k0);
#pragma unroll
        for (int tj = 0; tj < 4; ++tj)
          acc[ti][tj] = wmma_bf16(a, bfr[tj], acc[ti][tj]);
      }
    }
  }

  int mv[4];  // pad mask: one key column per lane per tj-tile
#pragma unroll
  for (int tj = 0; tj < 4; ++tj) mv[tj] = mask[b * SQ + k0c + tj * 16 + l15];

  float* __restrict__ Er = Eo + (size_t)b * SQ * SQ;  // uniform base
  const int rsel = hi16 * 8;
  const float negs = NEGV * INV_SCALE;
#pragma unroll
  for (int ti = 0; ti < 4; ++ti)
#pragma unroll
    for (int tj = 0; tj < 4; ++tj)
#pragma unroll
      for (int j = 0; j < 8; ++j) {
        int q  = q0  + ti * 16 + j + rsel;
        int kk = k0c + tj * 16 + l15;
        bool keep = (kk <= q) && (mv[tj] != 0);
        Er[q * SQ + kk] = keep ? acc[ti][tj][j] * INV_SCALE : negs;
      }
}

// ---------------- row softmax over 2048 keys, fp32 in place -----------------
__global__ void __launch_bounds__(256)
softmax_kernel(float* __restrict__ P) {
  __shared__ float red[256];
  const int t = threadIdx.x;
  float* p = P + (size_t)blockIdx.x * SQ;
  float v[8];
  float m = -3.402823466e38f;
#pragma unroll
  for (int i = 0; i < 8; ++i) { v[i] = p[t + 256 * i]; m = fmaxf(m, v[i]); }
  red[t] = m; __syncthreads();
  for (int s = 128; s > 0; s >>= 1) {
    if (t < s) red[t] = fmaxf(red[t], red[t + s]);
    __syncthreads();
  }
  const float rmax = red[0];
  __syncthreads();
  float sum = 0.f;
#pragma unroll
  for (int i = 0; i < 8; ++i) { v[i] = __expf(v[i] - rmax); sum += v[i]; }
  red[t] = sum; __syncthreads();
  for (int s = 128; s > 0; s >>= 1) {
    if (t < s) red[t] += red[t + s];
    __syncthreads();
  }
  const float inv = 1.0f / red[0];
#pragma unroll
  for (int i = 0; i < 8; ++i) p[t + 256 * i] = v[i] * inv;
}

// ---------------- output = P @ V  (A: fp32 attn -> bf16 in-register) --------
__global__ void __launch_bounds__(256)
av_kernel(const float* __restrict__ P,
          const bf16_t* __restrict__ VT,  // [BZ][DD][SQ]
          float* __restrict__ Out) {
  const int lane = threadIdx.x;
  const int b = blockIdx.z;
  const int q0 = (blockIdx.y * 8 + threadIdx.y) * 64;
  const int d0 = blockIdx.x * 64;
  const int l15 = lane & 15;
  const int hi16 = (lane >> 4) & 1;
  const float*  __restrict__ Pb = P  + (size_t)b * SQ * SQ;   // uniform bases
  const bf16_t* __restrict__ Vb = VT + (size_t)b * DD * SQ;
  const int kmax = q0 + 64;  // P is exactly 0 above the causal diagonal

  int aoffs[4], boffs[4];
#pragma unroll
  for (int t = 0; t < 4; ++t) {
    aoffs[t] = (q0 + t * 16 + l15) * SQ + hi16 * 8;
    boffs[t] = (d0 + t * 16 + l15) * SQ + hi16 * 8;
  }

  v8f acc[4][4] = {};
#pragma unroll 1
  for (int k0 = 0; k0 < kmax; k0 += 32) {
    v16bf bfr[4];
#pragma unroll
    for (int t = 0; t < 4; ++t) bfr[t] = load_b_frag(Vb, boffs[t] + k0);
#pragma unroll
    for (int ti = 0; ti < 4; ++ti) {
      const float* pr = Pb + aoffs[ti] + k0;  // fp32 attn -> bf16 in-register
      v4f x0 = *(const v4f*)(pr);
      v4f x1 = *(const v4f*)(pr + 4);
      v4f x2 = *(const v4f*)(pr + 16);
      v4f x3 = *(const v4f*)(pr + 20);
      v16bf a;
#pragma unroll
      for (int i = 0; i < 4; ++i) {
        a[i]      = (bf16_t)x0[i];
        a[4 + i]  = (bf16_t)x1[i];
        a[8 + i]  = (bf16_t)x2[i];
        a[12 + i] = (bf16_t)x3[i];
      }
#pragma unroll
      for (int tj = 0; tj < 4; ++tj)
        acc[ti][tj] = wmma_bf16(a, bfr[tj], acc[ti][tj]);
    }
  }

  float* __restrict__ Ob = Out + (size_t)b * SQ * DD;
  const int rsel = hi16 * 8;
#pragma unroll
  for (int ti = 0; ti < 4; ++ti)
#pragma unroll
    for (int tj = 0; tj < 4; ++tj)
#pragma unroll
      for (int j = 0; j < 8; ++j) {
        int q = q0 + ti * 16 + j + rsel;
        int d = d0 + tj * 16 + l15;
        Ob[q * DD + d] = acc[ti][tj][j];
      }
}

// ---------------- host-side launch ------------------------------------------
extern "C" void kernel_launch(void* const* d_in, const int* in_sizes, int n_in,
                              void* d_out, int out_size, void* d_ws, size_t ws_size,
                              hipStream_t stream) {
  (void)in_sizes; (void)n_in; (void)out_size; (void)ws_size;
  const float* emb  = (const float*)d_in[0];   // [BZ,SQ,ED]
  const int*   mask = (const int*)d_in[1];     // [BZ,SQ]
  const float* Wq   = (const float*)d_in[2];   // [ED,DD]
  const float* Wk   = (const float*)d_in[3];
  const float* Wv   = (const float*)d_in[4];

  float* out  = (float*)d_out;                   // [BZ,SQ,DD]
  float* attn = out + (size_t)BZ * SQ * DD;      // [BZ,SQ,SQ]

  // workspace carve-out (134 MB total, all 256B aligned power-of-two chunks)
  char* ws = (char*)d_ws;
  size_t off = 0;
  auto carve = [&](size_t bytes) -> char* {
    char* p = ws + off;
    off += (bytes + 255) & ~(size_t)255;
    return p;
  };
  bf16_t* embb = (bf16_t*)carve((size_t)BZ * SQ * ED * sizeof(bf16_t)); // 32 MB
  bf16_t* wqT  = (bf16_t*)carve((size_t)ED * DD * sizeof(bf16_t));      // 2 MB
  bf16_t* wkT  = (bf16_t*)carve((size_t)ED * DD * sizeof(bf16_t));
  bf16_t* wvT  = (bf16_t*)carve((size_t)ED * DD * sizeof(bf16_t));
  bf16_t* Qb   = (bf16_t*)carve((size_t)BZ * SQ * DD * sizeof(bf16_t)); // 32 MB
  bf16_t* Kb   = (bf16_t*)carve((size_t)BZ * SQ * DD * sizeof(bf16_t));
  bf16_t* VTb  = (bf16_t*)carve((size_t)BZ * DD * SQ * sizeof(bf16_t));

  // 1) precision conversion
  {
    int n = BZ * SQ * ED;
    cvt_bf16_kernel<<<n / 4 / 256, 256, 0, stream>>>(emb, embb, n);
    int nw = ED * DD, blk = (nw + 255) / 256;
    transpose_w_kernel<<<blk, 256, 0, stream>>>(Wq, wqT);
    transpose_w_kernel<<<blk, 256, 0, stream>>>(Wk, wkT);
    transpose_w_kernel<<<blk, 256, 0, stream>>>(Wv, wvT);
  }

  dim3 blk(32, 8);  // 8 wave32 waves per block, one 64x64 tile each

  // 2) QKV projections (V stored transposed per batch)
  dim3 gp(DD / 64, (BZ * SQ) / 64 / 8);
  proj_gemm_kernel<false><<<gp, blk, 0, stream>>>(embb, wqT, Qb);
  proj_gemm_kernel<false><<<gp, blk, 0, stream>>>(embb, wkT, Kb);
  proj_gemm_kernel<true ><<<gp, blk, 0, stream>>>(embb, wvT, VTb);

  // 3) masked scaled logits -> attention region of d_out
  energy_kernel<<<dim3(SQ / 64, SQ / 64 / 8, BZ), blk, 0, stream>>>(Qb, Kb, mask, attn);

  // 4) row softmax (fp32, in place)
  softmax_kernel<<<dim3(BZ * SQ), 256, 0, stream>>>(attn);

  // 5) output = attention @ V
  av_kernel<<<dim3(DD / 64, SQ / 64 / 8, BZ), blk, 0, stream>>>(attn, VTb, out);
}
